// MultiHeadAttentionBlock_53669911331251
// MI455X (gfx1250) — compile-verified, matrix operand order checked by reading, not run on hardware
//
#include <hip/hip_runtime.h>
#include <hip/hip_bf16.h>

#define D_MODEL   1024
#define NUM_HEADS 16
#define HEAD_DIM  64
#define BATCH     2
#define SEQ       2048

typedef __attribute__((ext_vector_type(16))) __bf16 v16bf;
typedef __attribute__((ext_vector_type(8)))  __bf16 v8bf;
typedef __attribute__((ext_vector_type(8)))  float  v8f;

__device__ __forceinline__ v8f wmma_bf16(v16bf a, v16bf b, v8f c) {
  return __builtin_amdgcn_wmma_f32_16x16x32_bf16(false, a, false, b, (short)0, c,
                                                 false, false);
}

// lane <-> lane^16 swap. VALU permlane if available, else shuffle (ds path).
__device__ __forceinline__ float xor16f(float x) {
#if __has_builtin(__builtin_amdgcn_permlanex16)
  int v = __builtin_amdgcn_permlanex16(__float_as_int(x), __float_as_int(x),
                                       0x76543210, (int)0xfedcba98, false, false);
  return __int_as_float(v);
#else
  return __shfl_xor(x, 16, 32);
#endif
}

// Fragment loader from fp32 memory (converts to bf16).
// p -> (row base + kb). Elements 0..7 <- p[0..7], 8..15 <- p[16..23]
__device__ __forceinline__ v16bf load_frag_f32(const float* __restrict__ p) {
  float4 x0 = *(const float4*)(p + 0);
  float4 x1 = *(const float4*)(p + 4);
  float4 x2 = *(const float4*)(p + 16);
  float4 x3 = *(const float4*)(p + 20);
  v16bf r;
  r[0]  = (__bf16)x0.x; r[1]  = (__bf16)x0.y; r[2]  = (__bf16)x0.z; r[3]  = (__bf16)x0.w;
  r[4]  = (__bf16)x1.x; r[5]  = (__bf16)x1.y; r[6]  = (__bf16)x1.z; r[7]  = (__bf16)x1.w;
  r[8]  = (__bf16)x2.x; r[9]  = (__bf16)x2.y; r[10] = (__bf16)x2.z; r[11] = (__bf16)x2.w;
  r[12] = (__bf16)x3.x; r[13] = (__bf16)x3.y; r[14] = (__bf16)x3.z; r[15] = (__bf16)x3.w;
  return r;
}

__device__ __forceinline__ v16bf load_frag_bf16(const __bf16* __restrict__ p) {
  v8bf lo = *(const v8bf*)(p);
  v8bf hi = *(const v8bf*)(p + 16);
  v16bf r;
#pragma unroll
  for (int i = 0; i < 8; ++i) { r[i] = lo[i]; r[8 + i] = hi[i]; }
  return r;
}

__device__ __forceinline__ void store8_bf16(__bf16* __restrict__ p, v8f a) {
  v8bf t;
#pragma unroll
  for (int i = 0; i < 8; ++i) t[i] = (__bf16)a[i];
  *(v8bf*)p = t;
}

__device__ __forceinline__ v8bf pack8(float4 a, float4 b) {
  v8bf t;
  t[0] = (__bf16)a.x; t[1] = (__bf16)a.y; t[2] = (__bf16)a.z; t[3] = (__bf16)a.w;
  t[4] = (__bf16)b.x; t[5] = (__bf16)b.y; t[6] = (__bf16)b.z; t[7] = (__bf16)b.w;
  return t;
}

// ---------------------------------------------------------------------------
// Projection GEMM: Y = cast_bf16( X[4096,1024] @ W^T ).
// WG = 256 thr = 8 waves; WG tile 128m x 128n; wave tile 16m x 128n.
// 128x32 weight tile staged fp32->bf16 in LDS, double buffered; next-tile
// global loads issued before the current WMMAs.
// ---------------------------------------------------------------------------
template <bool TRANS_V>
__global__ __launch_bounds__(256) void proj_kernel(const float* __restrict__ X,
                                                   const float* __restrict__ W,
                                                   __bf16* __restrict__ Y) {
  __shared__ __align__(16) __bf16 Bs[2][128][32];

  const int t    = threadIdx.x;
  const int lane = t & 31;
  const int wave = t >> 5;
  const int lh   = lane >> 4;
  const int l16  = lane & 15;
  const int kb   = lh * 8;

  const int m0 = blockIdx.x * 128 + wave * 16;  // row = b*S+s
  const int n0 = blockIdx.y * 128;              // col

  // staging: 2 threads per weight row, 16 k-values each
  const int srow = t >> 1;         // 0..127
  const int sks  = (t & 1) * 16;   // 0 or 16
  const float* wsrc = W + (size_t)(n0 + srow) * D_MODEL + sks;
  const float* arow = X + (size_t)(m0 + l16) * D_MODEL + kb;

  v8f acc[8];
#pragma unroll
  for (int g = 0; g < 8; ++g)
#pragma unroll
    for (int r = 0; r < 8; ++r) acc[g][r] = 0.0f;

  {  // stage k-tile 0
    float4 w0 = *(const float4*)(wsrc + 0);
    float4 w1 = *(const float4*)(wsrc + 4);
    float4 w2 = *(const float4*)(wsrc + 8);
    float4 w3 = *(const float4*)(wsrc + 12);
    *(v8bf*)&Bs[0][srow][sks]     = pack8(w0, w1);
    *(v8bf*)&Bs[0][srow][sks + 8] = pack8(w2, w3);
  }
  v16bf a_cur = load_frag_f32(arow);
  __syncthreads();

  int buf = 0;
  for (int kk = 0; kk < D_MODEL; kk += 32) {
    const bool has_next = (kk + 32) < D_MODEL;

    // issue next-tile global loads first (overlap with WMMAs)
    float4 w0{}, w1{}, w2{}, w3{};
    v16bf a_next = {};
    if (has_next) {
      w0 = *(const float4*)(wsrc + kk + 32);
      w1 = *(const float4*)(wsrc + kk + 36);
      w2 = *(const float4*)(wsrc + kk + 40);
      w3 = *(const float4*)(wsrc + kk + 44);
      a_next = load_frag_f32(arow + kk + 32);
    }
    __builtin_prefetch(arow + (kk + 64 < D_MODEL ? kk + 64 : kk), 0, 1);

#pragma unroll
    for (int g = 0; g < 8; ++g) {
      v8bf lo = *(const v8bf*)&Bs[buf][g * 16 + l16][kb];
      v8bf hi = *(const v8bf*)&Bs[buf][g * 16 + l16][kb + 16];
      v16bf bf_;
#pragma unroll
      for (int i = 0; i < 8; ++i) { bf_[i] = lo[i]; bf_[8 + i] = hi[i]; }
      acc[g] = wmma_bf16(a_cur, bf_, acc[g]);
    }

    if (has_next) {
      *(v8bf*)&Bs[buf ^ 1][srow][sks]     = pack8(w0, w1);
      *(v8bf*)&Bs[buf ^ 1][srow][sks + 8] = pack8(w2, w3);
      a_cur = a_next;
    }
    __syncthreads();
    buf ^= 1;
  }

  if (!TRANS_V) {
#pragma unroll
    for (int g = 0; g < 8; ++g) {
      const int n = n0 + g * 16 + l16;
#pragma unroll
      for (int r = 0; r < 8; ++r)
        Y[(size_t)(m0 + 8 * lh + r) * D_MODEL + n] = (__bf16)acc[g][r];
    }
  } else {
    const int b      = m0 / SEQ;
    const int s_base = (m0 % SEQ) + 8 * lh;
#pragma unroll
    for (int g = 0; g < 8; ++g) {
      const int n  = n0 + g * 16 + l16;
      const int h  = n >> 6;
      const int hd = n & 63;
      __bf16* dst = Y + (((size_t)(b * NUM_HEADS + h) * HEAD_DIM + hd) * SEQ) + s_base;
      store8_bf16(dst, acc[g]);
    }
  }
}

// ---------------------------------------------------------------------------
// Flash attention helpers
// ---------------------------------------------------------------------------
struct Frag4 { v16bf a00, a01, a10, a11; };

__device__ __forceinline__ Frag4 load_kfrags(const __bf16* __restrict__ Kbase,
                                             int j, int l16, int kb) {
  const __bf16* k0p = Kbase + (size_t)(j + l16) * D_MODEL + kb;
  const __bf16* k1p = k0p + (size_t)16 * D_MODEL;
  Frag4 f;
  f.a00 = load_frag_bf16(k0p);
  f.a01 = load_frag_bf16(k0p + 32);
  f.a10 = load_frag_bf16(k1p);
  f.a11 = load_frag_bf16(k1p + 32);
  return f;
}

// Online-softmax update for one 16-query tile (scores for 32 keys in st0/st1).
__device__ __forceinline__ float softmax_update(v8f& st0, v8f& st1,
                                                float& m_run, float& l_run) {
  float tmax = -3.0e38f;
#pragma unroll
  for (int r = 0; r < 8; ++r) tmax = fmaxf(tmax, fmaxf(st0[r], st1[r]));
  tmax = fmaxf(tmax, xor16f(tmax));  // lanes L / L+16 share a query

  const float m_new = fmaxf(m_run, tmax);
  const float corr  = __expf(m_run - m_new);

  float rsum = 0.0f;
#pragma unroll
  for (int r = 0; r < 8; ++r) {
    const float p0 = __expf(st0[r] - m_new);
    const float p1 = __expf(st1[r] - m_new);
    st0[r] = p0; st1[r] = p1;
    rsum += p0 + p1;
  }
  rsum += xor16f(rsum);
  l_run = l_run * corr + rsum;
  m_run = m_new;
  return corr;
}

// One 32-key block for TWO 16-query tiles sharing the K and V fragments.
template <bool MASKED>
__device__ __forceinline__ void attn_step(const Frag4& kf, int j, int qA,
                                          int kb, int l16,
                                          const v16bf& qa0, const v16bf& qa1,
                                          const v16bf& qc0, const v16bf& qc1,
                                          const __bf16* __restrict__ Vbase,
                                          v8f oA[4], v8f oB[4],
                                          float& mA, float& lA,
                                          float& mB, float& lB) {
  // V^T fragment loads issued early: latency hides under score WMMAs + softmax
  v16bf va[4];
#pragma unroll
  for (int g = 0; g < 4; ++g)
    va[g] = load_frag_bf16(Vbase + (size_t)(g * 16 + l16) * SEQ + j + kb);

  v8f z;
#pragma unroll
  for (int r = 0; r < 8; ++r) z[r] = 0.0f;

  // S^T = K_tile x Q^T for both query tiles (lane n = query, VGPR r = key)
  v8f sA0 = wmma_bf16(kf.a00, qa0, z);
  sA0     = wmma_bf16(kf.a01, qa1, sA0);
  v8f sA1 = wmma_bf16(kf.a10, qa0, z);
  sA1     = wmma_bf16(kf.a11, qa1, sA1);
  v8f sB0 = wmma_bf16(kf.a00, qc0, z);
  sB0     = wmma_bf16(kf.a01, qc1, sB0);
  v8f sB1 = wmma_bf16(kf.a10, qc0, z);
  sB1     = wmma_bf16(kf.a11, qc1, sB1);

  if (MASKED) {
    const int qB = qA + 16;
#pragma unroll
    for (int r = 0; r < 8; ++r) {
      const int key0 = j + kb + r;
      const int key1 = key0 + 16;
      sA0[r] = (key0 > qA) ? -3.0e38f : sA0[r];
      sA1[r] = (key1 > qA) ? -3.0e38f : sA1[r];
      sB0[r] = (key0 > qB) ? -3.0e38f : sB0[r];
      sB1[r] = (key1 > qB) ? -3.0e38f : sB1[r];
    }
  }

  const float corrA = softmax_update(sA0, sA1, mA, lA);
  const float corrB = softmax_update(sB0, sB1, mB, lB);

#pragma unroll
  for (int g = 0; g < 4; ++g)
#pragma unroll
    for (int r = 0; r < 8; ++r) { oA[g][r] *= corrA; oB[g][r] *= corrB; }

  // P^T is directly the B-fragment of O^T = V^T x P^T: register-only cast
  v16bf pbA, pbB;
#pragma unroll
  for (int r = 0; r < 8; ++r) {
    pbA[r] = (__bf16)sA0[r]; pbA[8 + r] = (__bf16)sA1[r];
    pbB[r] = (__bf16)sB0[r]; pbB[8 + r] = (__bf16)sB1[r];
  }

#pragma unroll
  for (int g = 0; g < 4; ++g) {
    oA[g] = wmma_bf16(va[g], pbA, oA[g]);
    oB[g] = wmma_bf16(va[g], pbB, oB[g]);
  }
}

__device__ __forceinline__ v16bf scale_frag(v16bf v) {
#pragma unroll
  for (int i = 0; i < 16; ++i) v[i] = (__bf16)((float)v[i] * 0.125f);
  return v;
}

// ---------------------------------------------------------------------------
// Flash attention (causal). Per wave: 32 queries (two 16-q tiles sharing all
// K/V fragments). Main loop unrolled 2x with ping-pong K fragments: every
// fragment assignment is a fresh global load (no register copies). Exactly
// one masked diagonal block at the tail.
// ---------------------------------------------------------------------------
__global__ __launch_bounds__(256) void attn_kernel(const __bf16* __restrict__ Q,
                                                   const __bf16* __restrict__ K,
                                                   const __bf16* __restrict__ Vt,
                                                   __bf16* __restrict__ O) {
  const int lane = threadIdx.x & 31;
  const int wave = threadIdx.x >> 5;
  const int lh   = lane >> 4;
  const int l16  = lane & 15;
  const int kb   = lh * 8;

  const int b  = blockIdx.z;
  const int h  = blockIdx.y;
  const int q0 = blockIdx.x * 256 + wave * 32;
  const int qA = q0 + l16;

  const __bf16* qrowA = Q + ((size_t)(b * SEQ) + q0 + l16) * D_MODEL + h * HEAD_DIM;
  const __bf16* qrowB = qrowA + (size_t)16 * D_MODEL;
  const v16bf qa0 = scale_frag(load_frag_bf16(qrowA + kb));
  const v16bf qa1 = scale_frag(load_frag_bf16(qrowA + 32 + kb));
  const v16bf qc0 = scale_frag(load_frag_bf16(qrowB + kb));
  const v16bf qc1 = scale_frag(load_frag_bf16(qrowB + 32 + kb));

  const __bf16* Kbase = K + (size_t)(b * SEQ) * D_MODEL + h * HEAD_DIM;
  const __bf16* Vbase = Vt + (size_t)(b * NUM_HEADS + h) * HEAD_DIM * SEQ;

  v8f oA[4], oB[4];
#pragma unroll
  for (int g = 0; g < 4; ++g)
#pragma unroll
    for (int r = 0; r < 8; ++r) { oA[g][r] = 0.0f; oB[g][r] = 0.0f; }

  float mA = -3.0e38f, lA = 0.0f;
  float mB = -3.0e38f, lB = 0.0f;

  const int jtail = q0;  // start of the single masked diagonal block

  if (jtail > 0) {
    Frag4 f0 = load_kfrags(Kbase, 0, l16, kb);
    int j = 0;
    // 2x-unrolled ping-pong: f1 load hides under step(f0); f0 reload (a fresh
    // global load into the same registers, not a copy) hides under step(f1).
    for (; j + 64 <= jtail; j += 64) {
      Frag4 f1 = load_kfrags(Kbase, j + 32, l16, kb);
      {  // L2 prefetch ~3 blocks ahead (clamped -> always valid, no branch)
        const int jp = (j + 96 < jtail) ? (j + 96) : jtail;
        __builtin_prefetch(Kbase + (size_t)(jp + l16) * D_MODEL + kb, 0, 1);
        __builtin_prefetch(Kbase + (size_t)(jp + 16 + l16) * D_MODEL + kb, 0, 1);
      }
      attn_step<false>(f0, j, qA, kb, l16, qa0, qa1, qc0, qc1, Vbase,
                       oA, oB, mA, lA, mB, lB);
      f0 = load_kfrags(Kbase, j + 64, l16, kb);  // safe: <= diagonal block rows
      attn_step<false>(f1, j + 32, qA, kb, l16, qa0, qa1, qc0, qc1, Vbase,
                       oA, oB, mA, lA, mB, lB);
    }
    if (j < jtail) {  // odd remainder: f0 already holds block j
      attn_step<false>(f0, j, qA, kb, l16, qa0, qa1, qc0, qc1, Vbase,
                       oA, oB, mA, lA, mB, lB);
    }
  }
  {  // masked diagonal block
    Frag4 ft = load_kfrags(Kbase, jtail, l16, kb);
    attn_step<true>(ft, jtail, qA, kb, l16, qa0, qa1, qc0, qc1, Vbase,
                    oA, oB, mA, lA, mB, lB);
  }

  const float invA = 1.0f / lA;
  const float invB = 1.0f / lB;
  __bf16* orowA = O + ((size_t)(b * SEQ) + qA) * D_MODEL + h * HEAD_DIM;
  __bf16* orowB = orowA + (size_t)16 * D_MODEL;
#pragma unroll
  for (int g = 0; g < 4; ++g) {
    v8f ta, tb;
#pragma unroll
    for (int r = 0; r < 8; ++r) { ta[r] = oA[g][r] * invA; tb[r] = oB[g][r] * invB; }
    store8_bf16(orowA + g * 16 + 8 * lh, ta);
    store8_bf16(orowB + g * 16 + 8 * lh, tb);
  }
}

// ---------------------------------------------------------------------------
// Output projection: out_f32 = X_bf16 @ Wo^T, same LDS-staged pipeline.
// ---------------------------------------------------------------------------
__global__ __launch_bounds__(256) void oproj_kernel(const __bf16* __restrict__ X,
                                                    const float* __restrict__ W,
                                                    float* __restrict__ Y) {
  __shared__ __align__(16) __bf16 Bs[2][128][32];

  const int t    = threadIdx.x;
  const int lane = t & 31;
  const int wave = t >> 5;
  const int lh   = lane >> 4;
  const int l16  = lane & 15;
  const int kb   = lh * 8;

  const int m0 = blockIdx.x * 128 + wave * 16;
  const int n0 = blockIdx.y * 128;

  const int srow = t >> 1;
  const int sks  = (t & 1) * 16;
  const float* wsrc  = W + (size_t)(n0 + srow) * D_MODEL + sks;
  const __bf16* arow = X + (size_t)(m0 + l16) * D_MODEL + kb;

  v8f acc[8];
#pragma unroll
  for (int g = 0; g < 8; ++g)
#pragma unroll
    for (int r = 0; r < 8; ++r) acc[g][r] = 0.0f;

  {
    float4 w0 = *(const float4*)(wsrc + 0);
    float4 w1 = *(const float4*)(wsrc + 4);
    float4 w2 = *(const float4*)(wsrc + 8);
    float4 w3 = *(const float4*)(wsrc + 12);
    *(v8bf*)&Bs[0][srow][sks]     = pack8(w0, w1);
    *(v8bf*)&Bs[0][srow][sks + 8] = pack8(w2, w3);
  }
  v16bf a_cur = load_frag_bf16(arow);
  __syncthreads();

  int buf = 0;
  for (int kk = 0; kk < D_MODEL; kk += 32) {
    const bool has_next = (kk + 32) < D_MODEL;

    float4 w0{}, w1{}, w2{}, w3{};
    v16bf a_next = {};
    if (has_next) {
      w0 = *(const float4*)(wsrc + kk + 32);
      w1 = *(const float4*)(wsrc + kk + 36);
      w2 = *(const float4*)(wsrc + kk + 40);
      w3 = *(const float4*)(wsrc + kk + 44);
      a_next = load_frag_bf16(arow + kk + 32);
    }
    __builtin_prefetch(arow + (kk + 64 < D_MODEL ? kk + 64 : kk), 0, 1);

#pragma unroll
    for (int g = 0; g < 8; ++g) {
      v8bf lo = *(const v8bf*)&Bs[buf][g * 16 + l16][kb];
      v8bf hi = *(const v8bf*)&Bs[buf][g * 16 + l16][kb + 16];
      v16bf bf_;
#pragma unroll
      for (int i = 0; i < 8; ++i) { bf_[i] = lo[i]; bf_[8 + i] = hi[i]; }
      acc[g] = wmma_bf16(a_cur, bf_, acc[g]);
    }

    if (has_next) {
      *(v8bf*)&Bs[buf ^ 1][srow][sks]     = pack8(w0, w1);
      *(v8bf*)&Bs[buf ^ 1][srow][sks + 8] = pack8(w2, w3);
      a_cur = a_next;
    }
    __syncthreads();
    buf ^= 1;
  }

#pragma unroll
  for (int g = 0; g < 8; ++g) {
    const int n = n0 + g * 16 + l16;
#pragma unroll
    for (int r = 0; r < 8; ++r)
      Y[(size_t)(m0 + 8 * lh + r) * D_MODEL + n] = acc[g][r];
  }
}

// ---------------------------------------------------------------------------
extern "C" void kernel_launch(void* const* d_in, const int* in_sizes, int n_in,
                              void* d_out, int out_size, void* d_ws, size_t ws_size,
                              hipStream_t stream) {
  const float* q  = (const float*)d_in[0];
  const float* k  = (const float*)d_in[1];
  const float* v  = (const float*)d_in[2];
  // d_in[3] = causal mask (triu bool) — applied analytically in attn_kernel.
  const float* Wq = (const float*)d_in[4];
  const float* Wk = (const float*)d_in[5];
  const float* Wv = (const float*)d_in[6];
  const float* Wo = (const float*)d_in[7];
  float* out = (float*)d_out;

  const size_t NELEM = (size_t)BATCH * SEQ * D_MODEL;  // 4M elems
  __bf16* Qb = (__bf16*)d_ws;
  __bf16* Kb = Qb + NELEM;
  __bf16* Vt = Qb + 2 * NELEM;
  __bf16* Xa = Qb + 3 * NELEM;

  dim3 blk(256);
  dim3 gproj((BATCH * SEQ) / 128, D_MODEL / 128);  // 32 x 8
  dim3 gattn(SEQ / 256, NUM_HEADS, BATCH);         // 8 x 16 x 2

  proj_kernel<false><<<gproj, blk, 0, stream>>>(q, Wq, Qb);
  proj_kernel<false><<<gproj, blk, 0, stream>>>(k, Wk, Kb);
  proj_kernel<true ><<<gproj, blk, 0, stream>>>(v, Wv, Vt);
  attn_kernel<<<gattn, blk, 0, stream>>>(Qb, Kb, Vt, Xa);
  oproj_kernel<<<gproj, blk, 0, stream>>>(Xa, Wo, out);
}
